// EnhancedGNNModel_21629455302740
// MI455X (gfx1250) — matrix operand order repaired and checked by reading, not compiled, not run on hardware
//
#include <hip/hip_runtime.h>
#include <hip/hip_bf16.h>
#include <stdint.h>

// ---------------------------------------------------------------------------
// EnhancedGNNModel on MI455X (gfx1250).  wave32, WMMA bf16 GEMMs with async
// LDS staging, atomic scatter for graph propagation.  H = F = 128 hardcoded.
// ---------------------------------------------------------------------------

typedef __attribute__((ext_vector_type(16))) __bf16 v16bf;
typedef __attribute__((ext_vector_type(8)))  float  v8f;
typedef __attribute__((ext_vector_type(4)))  int    v4i;

union ABfrag { v16bf v; uint4 q[2]; };

#define NEG_SLOPE 0.2f
#define BN_EPS    1e-5f

#if defined(__HIP_DEVICE_COMPILE__) && __has_builtin(__builtin_amdgcn_global_load_async_to_lds_b128)
#define HAVE_ASYNC_LDS 1
#else
#define HAVE_ASYNC_LDS 0
#endif

__device__ __forceinline__ void stage16(uint4* lds_dst, const uint4* g_src) {
#if HAVE_ASYNC_LDS
    __builtin_amdgcn_global_load_async_to_lds_b128(
        (__attribute__((address_space(1))) v4i*)g_src,
        (__attribute__((address_space(3))) v4i*)lds_dst, 0, 0);
#else
    *lds_dst = *g_src;
#endif
}

__device__ __forceinline__ float leakyf(float x) { return x > 0.f ? x : NEG_SLOPE * x; }

__device__ __forceinline__ float atomicMaxF(float* addr, float val) {
    int* ai = (int*)addr;
    int old = __float_as_int(*addr);
    while (__int_as_float(old) < val) {
        int assumed = old;
        old = atomicCAS(ai, assumed, __float_as_int(val));
        if (old == assumed) break;
    }
    return __int_as_float(old);
}

// ------------------------------ utility ------------------------------------

__global__ void k_fill(float* p, float v, int n) {
    int t = blockIdx.x * blockDim.x + threadIdx.x;
    if (t < n) p[t] = v;
}

__global__ void k_f32_to_bf16(const float* __restrict__ src, __bf16* __restrict__ dst, int n) {
    int t = blockIdx.x * blockDim.x + threadIdx.x;
    if (t < n) dst[t] = (__bf16)src[t];
}

// transpose-convert a 128x128 f32 weight to bf16:  dst[c*128+r] = src[r*128+c]
__global__ void k_f32_to_bf16_T(const float* __restrict__ src, __bf16* __restrict__ dst) {
    int t = blockIdx.x * blockDim.x + threadIdx.x;
    if (t < 128 * 128) {
        int r = t >> 7, c = t & 127;
        dst[c * 128 + r] = (__bf16)src[t];
    }
}

// ------------------------------ WMMA GEMM ----------------------------------
// C[M x 128] = A[M x 128] * B[128 x 128]   (+bias, optional relu)
// BT = B transposed (128x128 bf16, BT[n][k]).  One block = 64 rows (4 M tiles);
// 8 waves own the 8 N tiles and loop the 4 M tiles, reusing B fragments 4x.
__global__ __launch_bounds__(256) void k_gemm128(const __bf16* __restrict__ A,
                                                 const __bf16* __restrict__ BT,
                                                 float* __restrict__ C, int M,
                                                 const float* __restrict__ bias,
                                                 int relu) {
    __shared__ __align__(16) __bf16 As[64 * 128];    // 16 KB
    __shared__ __align__(16) __bf16 Bs[128 * 128];   // 32 KB (transposed: [n][k])
    const int tid  = threadIdx.x;
    const int row0 = blockIdx.x * 64;

    // cooperative staging into LDS, 16B chunks (async on gfx1250 toolchains)
    for (int i = tid; i < 64 * 128 / 8; i += 256) {
        int r = i >> 4, c = i & 15;                  // 16 chunks per row
        int gr = row0 + r; if (gr >= M) gr = M - 1;
        stage16((uint4*)As + i, (const uint4*)(A + (size_t)gr * 128) + c);
    }
    for (int i = tid; i < 128 * 128 / 8; i += 256) {
        stage16((uint4*)Bs + i, (const uint4*)BT + i);
    }
#if HAVE_ASYNC_LDS
    asm volatile("s_wait_asynccnt 0x0" ::: "memory");
#endif
    __syncthreads();

    const int wave  = tid >> 5;      // ntile 0..7
    const int lane  = tid & 31;
    const int lidx  = lane & 15;
    const int lhalf = lane >> 4;
    const int ncol  = wave * 16 + lidx;
    const int kb    = lhalf * 16;

    v8f acc[4] = {};
    for (int k0 = 0; k0 < 128; k0 += 32) {
        // B fragment: lane = column n, elements e = K (k0 + lhalf*16 + e), contiguous
        ABfrag b;
        const __bf16* bp = &Bs[ncol * 128 + k0 + lhalf * 16];
        b.q[0] = *(const uint4*)(bp);
        b.q[1] = *(const uint4*)(bp + 8);
        #pragma unroll
        for (int r = 0; r < 4; r++) {
            // A fragment: lane = row (lidx), K runs {kb..kb+15}, contiguous
            ABfrag a;
            const __bf16* ap = &As[(r * 16 + lidx) * 128 + k0 + kb];
            a.q[0] = *(const uint4*)(ap);
            a.q[1] = *(const uint4*)(ap + 8);
            acc[r] = __builtin_amdgcn_wmma_f32_16x16x32_bf16(false, a.v, false, b.v,
                                                             (short)0, acc[r], false, false);
        }
    }

    const float bv = bias ? bias[ncol] : 0.0f;
    #pragma unroll
    for (int r = 0; r < 4; r++) {
        #pragma unroll
        for (int j = 0; j < 8; j++) {
            int row = row0 + r * 16 + lhalf * 8 + j;  // D: lanes 0-15 rows 0-7, 16-31 rows 8-15
            if (row < M) {
                float v = acc[r][j] + bv;
                if (relu) v = fmaxf(v, 0.0f);
                C[(size_t)row * 128 + ncol] = v;
            }
        }
    }
}

// ------------------------------ GCN ----------------------------------------

__global__ void k_deg_edges(const int* __restrict__ dst, float* deg, int E) {
    int t = blockIdx.x * blockDim.x + threadIdx.x;
    if (t < E) atomicAdd(&deg[dst[t]], 1.0f);
}

__global__ void k_deg_to_dis(float* d, int n) {
    int t = blockIdx.x * blockDim.x + threadIdx.x;
    if (t < n) d[t] = rsqrtf(fmaxf(d[t], 1.0f));   // deg >= 1 (self loop)
}

// self-loop message initializes the accumulator: out[i] = h[i] * dis[i]^2
__global__ void k_gcn_self(const float* __restrict__ h, const float* __restrict__ dis,
                           float* __restrict__ out, int n) {
    int t = blockIdx.x * blockDim.x + threadIdx.x;
    if (t < n * 128) {
        float di = dis[t >> 7];
        out[t] = h[t] * di * di;
    }
}

// wave per edge, float4 per lane (32*4 = 128 features)
__global__ void k_gcn_edges(const int* __restrict__ src, const int* __restrict__ dst,
                            const float* __restrict__ h, const float* __restrict__ dis,
                            float* __restrict__ out, int E) {
    int idx = blockIdx.x * blockDim.x + threadIdx.x;
    int e = idx >> 5, lane = idx & 31;
    if (e >= E) return;
    int s = src[e], d = dst[e];
    float nrm = dis[s] * dis[d];
    float4 v = ((const float4*)(h + (size_t)s * 128))[lane];
    float* od = out + (size_t)d * 128 + lane * 4;
    atomicAdd(od + 0, v.x * nrm);
    atomicAdd(od + 1, v.y * nrm);
    atomicAdd(od + 2, v.z * nrm);
    atomicAdd(od + 3, v.w * nrm);
}

// ------------------------------ BatchNorm + ReLU ---------------------------

__global__ __launch_bounds__(256) void k_bn_stats(const float* __restrict__ h,
                                                  float* __restrict__ mv, int n) {
    int c = blockIdx.x;                    // one block per column
    __shared__ float ssum[256], ssq[256];
    float s = 0.f, q = 0.f;
    for (int r = threadIdx.x; r < n; r += 256) {
        float v = h[(size_t)r * 128 + c];
        s += v; q += v * v;
    }
    ssum[threadIdx.x] = s; ssq[threadIdx.x] = q;
    __syncthreads();
    for (int st = 128; st > 0; st >>= 1) {
        if (threadIdx.x < st) {
            ssum[threadIdx.x] += ssum[threadIdx.x + st];
            ssq[threadIdx.x]  += ssq[threadIdx.x + st];
        }
        __syncthreads();
    }
    if (threadIdx.x == 0) {
        float mean = ssum[0] / n;
        float var  = ssq[0] / n - mean * mean;
        mv[c]       = mean;
        mv[128 + c] = rsqrtf(var + BN_EPS);
    }
}

__global__ void k_bn_relu(float* __restrict__ h, const float* __restrict__ mv,
                          const float* __restrict__ g, const float* __restrict__ be, int n) {
    int t = blockIdx.x * blockDim.x + threadIdx.x;
    if (t < n * 128) {
        int c = t & 127;
        float v = g[c] * (h[t] - mv[c]) * mv[128 + c] + be[c];
        h[t] = fmaxf(v, 0.0f);
    }
}

// ------------------------------ GAT ----------------------------------------

__global__ void k_att_dots(const float* __restrict__ h, const float* __restrict__ asrc,
                           const float* __restrict__ adst, float* als, float* ald, int n) {
    int idx = blockIdx.x * blockDim.x + threadIdx.x;
    int node = idx >> 5, lane = idx & 31;
    if (node >= n) return;
    const float* hr = h + (size_t)node * 128;
    float s = 0.f, d = 0.f;
    for (int f = lane; f < 128; f += 32) { float hv = hr[f]; s += hv * asrc[f]; d += hv * adst[f]; }
    for (int off = 16; off > 0; off >>= 1) { s += __shfl_down(s, off); d += __shfl_down(d, off); }
    if (lane == 0) { als[node] = s; ald[node] = d; }
}

__global__ void k_gat_init_m(const float* als, const float* ald, float* m, int n) {
    int t = blockIdx.x * blockDim.x + threadIdx.x;
    if (t < n) m[t] = leakyf(als[t] + ald[t]);     // self-loop attention logit
}

__global__ void k_gat_edge_max(const int* src, const int* dst, const float* als,
                               const float* ald, float* m, int E) {
    int t = blockIdx.x * blockDim.x + threadIdx.x;
    if (t < E) atomicMaxF(&m[dst[t]], leakyf(als[src[t]] + ald[dst[t]]));
}

__global__ void k_gat_init_accum(const float* __restrict__ h, const float* als,
                                 const float* ald, const float* m, float* den,
                                 float* __restrict__ out, int n) {
    int idx = blockIdx.x * blockDim.x + threadIdx.x;
    int node = idx >> 5, lane = idx & 31;
    if (node >= n) return;
    float ex = __expf(leakyf(als[node] + ald[node]) - m[node]);
    if (lane == 0) den[node] = ex;
    float4 v = ((const float4*)(h + (size_t)node * 128))[lane];
    float4 r = make_float4(v.x * ex, v.y * ex, v.z * ex, v.w * ex);
    ((float4*)(out + (size_t)node * 128))[lane] = r;
}

__global__ void k_gat_edge_accum(const int* __restrict__ src, const int* __restrict__ dst,
                                 const float* __restrict__ h, const float* als,
                                 const float* ald, const float* m, float* den,
                                 float* __restrict__ out, int E) {
    int idx = blockIdx.x * blockDim.x + threadIdx.x;
    int e = idx >> 5, lane = idx & 31;
    if (e >= E) return;
    int s = src[e], d = dst[e];
    float ex = __expf(leakyf(als[s] + ald[d]) - m[d]);
    if (lane == 0) atomicAdd(&den[d], ex);
    float4 v = ((const float4*)(h + (size_t)s * 128))[lane];
    float* od = out + (size_t)d * 128 + lane * 4;
    atomicAdd(od + 0, v.x * ex);
    atomicAdd(od + 1, v.y * ex);
    atomicAdd(od + 2, v.z * ex);
    atomicAdd(od + 3, v.w * ex);
}

__global__ void k_gat_final(float* __restrict__ h, const float* den, int n) {
    int t = blockIdx.x * blockDim.x + threadIdx.x;
    if (t < n * 128) h[t] /= den[t >> 7];
}

// ------------------------------ pool + MLP ---------------------------------

__global__ void k_pool_accum(const float* __restrict__ h, const int* __restrict__ batch,
                             float* pooled, float* cnt, int n) {
    int idx = blockIdx.x * blockDim.x + threadIdx.x;
    int node = idx >> 5, lane = idx & 31;
    if (node >= n) return;
    int b = batch[node];
    if (lane == 0) atomicAdd(&cnt[b], 1.0f);
    float4 v = ((const float4*)(h + (size_t)node * 128))[lane];
    float* pd = pooled + (size_t)b * 128 + lane * 4;
    atomicAdd(pd + 0, v.x); atomicAdd(pd + 1, v.y);
    atomicAdd(pd + 2, v.z); atomicAdd(pd + 3, v.w);
}

__global__ void k_pool_div(float* pooled, const float* cnt, int G) {
    int t = blockIdx.x * blockDim.x + threadIdx.x;
    if (t < G * 128) pooled[t] /= fmaxf(cnt[t >> 7], 1.0f);
}

__global__ void k_mlp2(const float* __restrict__ z, const float* __restrict__ fw2,
                       const float* __restrict__ fb2, float* out, int G) {
    int idx = blockIdx.x * blockDim.x + threadIdx.x;
    int g = idx >> 5, lane = idx & 31;
    if (g >= G) return;
    const float* zr = z + (size_t)g * 128;
    float s = 0.f;
    for (int f = lane; f < 128; f += 32) s += zr[f] * fw2[f];
    for (int off = 16; off > 0; off >>= 1) s += __shfl_down(s, off);
    if (lane == 0) out[g] = s + fb2[0];
}

// ------------------------------ host ---------------------------------------

extern "C" void kernel_launch(void* const* d_in, const int* in_sizes, int n_in,
                              void* d_out, int out_size, void* d_ws, size_t ws_size,
                              hipStream_t stream) {
    const float* x    = (const float*)d_in[0];
    const int*   ei   = (const int*)d_in[1];
    const int*   batch= (const int*)d_in[2];
    const float* W1   = (const float*)d_in[3];
    // b1 (d_in[4]) absorbed exactly by following BatchNorm -> skipped.
    const float* g1   = (const float*)d_in[5];
    const float* be1  = (const float*)d_in[6];
    const float* Wg   = (const float*)d_in[7];
    const float* attS = (const float*)d_in[8];
    const float* attD = (const float*)d_in[9];
    // bg (d_in[10]) absorbed by BN -> skipped.
    const float* g2   = (const float*)d_in[11];
    const float* be2  = (const float*)d_in[12];
    const float* W3   = (const float*)d_in[13];
    // b3 (d_in[14]) absorbed by BN -> skipped.
    const float* g3   = (const float*)d_in[15];
    const float* be3  = (const float*)d_in[16];
    const float* fw1  = (const float*)d_in[17];
    const float* fb1  = (const float*)d_in[18];
    const float* fw2  = (const float*)d_in[19];
    const float* fb2  = (const float*)d_in[20];
    float* out = (float*)d_out;

    const int Nn = in_sizes[0] / 128;
    const int E  = in_sizes[1] / 2;
    const int G  = out_size;              // [G,1] output
    const int* src = ei;
    const int* dst = ei + E;

    // -------- workspace carve (256B aligned) --------
    char* ws = (char*)d_ws; size_t off = 0;
    auto carve = [&](size_t bytes) -> void* {
        void* p = ws + off;
        off = (off + bytes + 255) & ~(size_t)255;
        return p;
    };
    __bf16* HBF  = (__bf16*)carve((size_t)Nn * 128 * 2);
    __bf16* WTBF = (__bf16*)carve(128 * 128 * 2);     // transposed weight, bf16
    float*  HLIN = (float*) carve((size_t)Nn * 128 * 4);
    float*  HAGG = (float*) carve((size_t)Nn * 128 * 4);
    float*  DIS  = (float*) carve((size_t)Nn * 4);
    float*  ALS  = (float*) carve((size_t)Nn * 4);
    float*  ALD  = (float*) carve((size_t)Nn * 4);
    float*  MBUF = (float*) carve((size_t)Nn * 4);
    float*  DEN  = (float*) carve((size_t)Nn * 4);
    float*  MV   = (float*) carve(256 * 4);
    float*  POOL = (float*) carve((size_t)G * 128 * 4);
    float*  CNT  = (float*) carve((size_t)G * 4);
    __bf16* PBF  = (__bf16*)carve((size_t)G * 128 * 2);
    float*  Z    = (float*) carve((size_t)G * 128 * 4);
    (void)ws_size; (void)n_in;

    const int TB = 256;
    auto gr1  = [&](int n) { return dim3((n + TB - 1) / TB); };            // thread per elem
    auto gr32 = [&](int n) { return dim3(((size_t)n * 32 + TB - 1) / TB); };// wave per item
    auto grmm = [&](int m) { return dim3((m + 63) / 64); };                // 64 rows per block
    const dim3 blk(TB);
    const int NE = Nn * 128;

    // ---- degree / symmetric norm (shared across both GCN layers) ----
    k_fill<<<gr1(Nn), blk, 0, stream>>>(DIS, 1.0f, Nn);          // self loop
    k_deg_edges<<<gr1(E), blk, 0, stream>>>(dst, DIS, E);
    k_deg_to_dis<<<gr1(Nn), blk, 0, stream>>>(DIS, Nn);

    // ---- layer 1: GCN -> BN -> ReLU ----
    k_f32_to_bf16<<<gr1(NE), blk, 0, stream>>>(x, HBF, NE);
    k_f32_to_bf16_T<<<gr1(16384), blk, 0, stream>>>(W1, WTBF);
    k_gemm128<<<grmm(Nn), blk, 0, stream>>>(HBF, WTBF, HLIN, Nn, nullptr, 0);
    k_gcn_self<<<gr1(NE), blk, 0, stream>>>(HLIN, DIS, HAGG, Nn);
    k_gcn_edges<<<gr32(E), blk, 0, stream>>>(src, dst, HLIN, DIS, HAGG, E);
    k_bn_stats<<<dim3(128), blk, 0, stream>>>(HAGG, MV, Nn);
    k_bn_relu<<<gr1(NE), blk, 0, stream>>>(HAGG, MV, g1, be1, Nn);

    // ---- layer 2: GAT -> BN -> ReLU ----
    k_f32_to_bf16<<<gr1(NE), blk, 0, stream>>>(HAGG, HBF, NE);
    k_f32_to_bf16_T<<<gr1(16384), blk, 0, stream>>>(Wg, WTBF);
    k_gemm128<<<grmm(Nn), blk, 0, stream>>>(HBF, WTBF, HLIN, Nn, nullptr, 0);
    k_att_dots<<<gr32(Nn), blk, 0, stream>>>(HLIN, attS, attD, ALS, ALD, Nn);
    k_gat_init_m<<<gr1(Nn), blk, 0, stream>>>(ALS, ALD, MBUF, Nn);
    k_gat_edge_max<<<gr1(E), blk, 0, stream>>>(src, dst, ALS, ALD, MBUF, E);
    k_gat_init_accum<<<gr32(Nn), blk, 0, stream>>>(HLIN, ALS, ALD, MBUF, DEN, HAGG, Nn);
    k_gat_edge_accum<<<gr32(E), blk, 0, stream>>>(src, dst, HLIN, ALS, ALD, MBUF, DEN, HAGG, E);
    k_gat_final<<<gr1(NE), blk, 0, stream>>>(HAGG, DEN, Nn);
    k_bn_stats<<<dim3(128), blk, 0, stream>>>(HAGG, MV, Nn);
    k_bn_relu<<<gr1(NE), blk, 0, stream>>>(HAGG, MV, g2, be2, Nn);

    // ---- layer 3: GCN -> BN -> ReLU ----
    k_f32_to_bf16<<<gr1(NE), blk, 0, stream>>>(HAGG, HBF, NE);
    k_f32_to_bf16_T<<<gr1(16384), blk, 0, stream>>>(W3, WTBF);
    k_gemm128<<<grmm(Nn), blk, 0, stream>>>(HBF, WTBF, HLIN, Nn, nullptr, 0);
    k_gcn_self<<<gr1(NE), blk, 0, stream>>>(HLIN, DIS, HAGG, Nn);
    k_gcn_edges<<<gr32(E), blk, 0, stream>>>(src, dst, HLIN, DIS, HAGG, E);
    k_bn_stats<<<dim3(128), blk, 0, stream>>>(HAGG, MV, Nn);
    k_bn_relu<<<gr1(NE), blk, 0, stream>>>(HAGG, MV, g3, be3, Nn);

    // ---- global mean pool ----
    k_fill<<<gr1(G * 128), blk, 0, stream>>>(POOL, 0.0f, G * 128);
    k_fill<<<gr1(G), blk, 0, stream>>>(CNT, 0.0f, G);
    k_pool_accum<<<gr32(Nn), blk, 0, stream>>>(HAGG, batch, POOL, CNT, Nn);
    k_pool_div<<<gr1(G * 128), blk, 0, stream>>>(POOL, CNT, G);

    // ---- MLP head:  relu(pooled@fw1 + fb1) @ fw2 + fb2 ----
    k_f32_to_bf16<<<gr1(G * 128), blk, 0, stream>>>(POOL, PBF, G * 128);
    k_f32_to_bf16_T<<<gr1(16384), blk, 0, stream>>>(fw1, WTBF);
    k_gemm128<<<grmm(G), blk, 0, stream>>>(PBF, WTBF, Z, G, fb1, 1);
    k_mlp2<<<gr32(G), blk, 0, stream>>>(Z, fw2, fb2, out, G);
}
